// MultiHeadAttention_80350248173694
// MI455X (gfx1250) — matrix operand order repaired and checked by reading, not compiled, not run on hardware
//
#include <hip/hip_runtime.h>

// MultiHeadAttention forward for MI455X (gfx1250).
// bf16 WMMA (v_wmma_f32_16x16x32_bf16) + f32 accumulate; all GEMM tile staging
// via Tensor Data Mover (tensor_load_to_lds, TENSORcnt), double-buffered.

typedef __bf16 bf16_t;
typedef __attribute__((ext_vector_type(16))) __bf16 v16bf;
typedef __attribute__((ext_vector_type(8)))  float  v8f;
typedef __attribute__((ext_vector_type(4)))  unsigned int u32x4;
typedef __attribute__((ext_vector_type(8)))  int i32x8;
typedef __attribute__((ext_vector_type(4)))  int i32x4;

#define B_SZ   8
#define LSEQ   1024
#define DMODEL 1024
#define NHEAD  16
#define DK     64
#define MROWS  (B_SZ * LSEQ)   // 8192

__device__ __forceinline__ bf16_t f2bf(float f) {
  union { float f; unsigned u; } a; a.f = f;
  unsigned r = a.u + 0x7FFFu + ((a.u >> 16) & 1u);   // round-to-nearest-even
  unsigned short h = (unsigned short)(r >> 16);
  return __builtin_bit_cast(bf16_t, h);
}

union FragU { uint4 u[2]; v16bf v; };

// A fragment (16x32, MxK): lane l row m=l&15; chunks at k=(l>>4)*8 and +16.
__device__ __forceinline__ v16bf frag_a(const bf16_t* p) {
  FragU r;
  r.u[0] = *(const uint4*)(p);
  r.u[1] = *(const uint4*)(p + 16);
  return r.v;
}
// B fragment (32x16, KxN) staged [N][K]: lane l col n=l&15; chunks at
// k=(l>>4)*8 and +16.
__device__ __forceinline__ v16bf frag_b(const bf16_t* p) {
  FragU r;
  r.u[0] = *(const uint4*)(p);
  r.u[1] = *(const uint4*)(p + 16);
  return r.v;
}

#define WMMA_BF16(A, Bv, C) \
  __builtin_amdgcn_wmma_f32_16x16x32_bf16(false, (A), false, (Bv), (short)0, (C), false, false)

__device__ __forceinline__ unsigned lds_addr_of(const void* p) {
  return (unsigned)(unsigned long long)p;   // low 32 bits of flat addr = LDS byte addr
}

// TDM: 2D tile (bf16 elements) from global (row stride stride_d0 elems) to LDS.
// D# layout per CDNA5 ISA ch.8: group0 {count,lds,global,type=2},
// group1 {data_size=2B, tensor dims, tile dims, stride}.
__device__ __forceinline__ void tdm_load_2d_bf16(unsigned lds, const void* gptr,
                                                 unsigned tensor_d0, unsigned tensor_d1,
                                                 unsigned stride_d0,
                                                 unsigned tile_d0, unsigned tile_d1) {
  unsigned long long ga = (unsigned long long)gptr;
  u32x4 g0;
  g0[0] = 1u;                                   // count=1 valid descriptor
  g0[1] = lds;                                  // lds_addr
  g0[2] = (unsigned)ga;                         // global_addr[31:0]
  g0[3] = (unsigned)((ga >> 32) & 0x01FFFFFFu) | 0x80000000u;  // addr[56:32]|type=2
  i32x8 g1;
  g1[0] = 0x00010000;                           // data_size=1 (2 bytes)
  g1[1] = (int)((tensor_d0 & 0xFFFFu) << 16);   // tensor_dim0[15:0]
  g1[2] = (int)(((tensor_d0 >> 16) & 0xFFFFu) | ((tensor_d1 & 0xFFFFu) << 16));
  g1[3] = (int)(((tensor_d1 >> 16) & 0xFFFFu) | ((tile_d0 & 0xFFFFu) << 16));
  g1[4] = (int)(tile_d1 & 0xFFFFu);             // tile_dim1; tile_dim2=0
  g1[5] = (int)stride_d0;                       // tensor_dim0_stride[31:0]
  g1[6] = 0;
  g1[7] = 0;
  i32x4 z4 = {0, 0, 0, 0};
  i32x8 z8 = {0, 0, 0, 0, 0, 0, 0, 0};
  __builtin_amdgcn_tensor_load_to_lds(g0, g1, z4, z4, z8, 0);
}

#define WAIT_TENSOR(n) __builtin_amdgcn_s_wait_tensorcnt((short)(n))

__global__ __launch_bounds__(256) void cvt_f32_bf16(const float* __restrict__ in,
                                                    bf16_t* __restrict__ out, int n) {
  int i = blockIdx.x * blockDim.x + threadIdx.x;
  int stride = gridDim.x * blockDim.x;
  for (; i < n; i += stride) out[i] = f2bf(in[i]);
}

// Wt[n][k] = W[k][n] with f32->bf16, 1024x1024, LDS-tiled transpose.
__global__ __launch_bounds__(256) void transpose_cvt_kernel(const float* __restrict__ in,
                                                            bf16_t* __restrict__ out) {
  __shared__ float tile[32][33];
  const int bn = blockIdx.x * 32, bk = blockIdx.y * 32;
  const int tx = threadIdx.x & 31, ty = threadIdx.x >> 5;  // 32 x 8
  for (int r = 0; r < 32; r += 8)
    tile[ty + r][tx] = in[(size_t)(bk + ty + r) * DMODEL + bn + tx];
  __syncthreads();
  for (int r = 0; r < 32; r += 8)
    out[(size_t)(bn + ty + r) * DMODEL + bk + tx] = f2bf(tile[tx][ty + r]);
}

// C[M,N] = A[M,K] @ Wt[N,K]^T (+bias)*scale.  A,Wt bf16 row-major.
// vtrans: write out as vhT[b,h,d,l] instead of [m][n].
__global__ __launch_bounds__(256) void gemm_bf16_kernel(
    const bf16_t* __restrict__ A, const bf16_t* __restrict__ Wt,
    const float* __restrict__ bias, bf16_t* __restrict__ outB,
    float* __restrict__ outF, int M, int N, int K, float scale, int vtrans) {
  __shared__ __align__(16) bf16_t sA[2][128 * 32];
  __shared__ __align__(16) bf16_t sB[2][128 * 32];
  const int tid = threadIdx.x;
  const int lane = tid & 31, w = tid >> 5;
  const int wrow = w >> 2, wcol = w & 3;          // 2x4 waves, 64x32 each
  const int m0 = blockIdx.y * 128, n0 = blockIdx.x * 128;
  const bool issuer = (w == 0);

  v8f acc[4][2];
  const v8f vzero = {0.f, 0.f, 0.f, 0.f, 0.f, 0.f, 0.f, 0.f};
  for (int i = 0; i < 4; i++)
    for (int j = 0; j < 2; j++) acc[i][j] = vzero;

  const int nk = K >> 5;
  if (issuer) {
    tdm_load_2d_bf16(lds_addr_of(sA[0]), A + (size_t)m0 * K, K, M - m0, K, 32, 128);
    tdm_load_2d_bf16(lds_addr_of(sB[0]), Wt + (size_t)n0 * K, K, N - n0, K, 32, 128);
  }
  for (int ki = 0; ki < nk; ki++) {
    const int cur = ki & 1, nxt = cur ^ 1;
    if (issuer) {
      if (ki + 1 < nk) {
        tdm_load_2d_bf16(lds_addr_of(sA[nxt]), A + (size_t)m0 * K + (ki + 1) * 32,
                         K, M - m0, K, 32, 128);
        tdm_load_2d_bf16(lds_addr_of(sB[nxt]), Wt + (size_t)n0 * K + (ki + 1) * 32,
                         K, N - n0, K, 32, 128);
        WAIT_TENSOR(2);   // two oldest (tile ki) done; TDM completes in order
      } else {
        WAIT_TENSOR(0);
      }
    }
    __syncthreads();
    const bf16_t* aB = sA[cur] + (wrow * 64 + (lane & 15)) * 32 + ((lane >> 4) << 3);
    const bf16_t* bB = sB[cur] + (wcol * 32 + (lane & 15)) * 32 + ((lane >> 4) << 3);
    v16bf b0 = frag_b(bB);
    v16bf b1 = frag_b(bB + 16 * 32);
    for (int i = 0; i < 4; i++) {
      v16bf a = frag_a(aB + i * 16 * 32);
      acc[i][0] = WMMA_BF16(a, b0, acc[i][0]);
      acc[i][1] = WMMA_BF16(a, b1, acc[i][1]);
    }
    __syncthreads();
  }

  const int mrB = 8 * (lane >> 4);
  for (int i = 0; i < 4; i++)
    for (int j = 0; j < 2; j++) {
      int n = n0 + wcol * 32 + j * 16 + (lane & 15);
      float bv = bias ? bias[n] : 0.0f;
      for (int r = 0; r < 8; r++) {
        int m = m0 + wrow * 64 + i * 16 + mrB + r;
        float vv = (acc[i][j][r] + bv) * scale;
        if (outF) {
          outF[(size_t)m * N + n] = vv;
        } else if (vtrans) {
          // vhT[b][h][d][l]
          outB[(((size_t)(m >> 10) * NHEAD + (n >> 6)) * DK + (n & 63)) * LSEQ +
               (m & 1023)] = f2bf(vv);
        } else {
          outB[(size_t)m * N + n] = f2bf(vv);
        }
      }
    }
}

// scores[h*B+b, q, k] = sum_d qh[b,q,h*64+d] * kh[b,k,h*64+d]
__global__ __launch_bounds__(256) void attn_scores_kernel(
    const bf16_t* __restrict__ QH, const bf16_t* __restrict__ KH,
    float* __restrict__ P) {
  __shared__ __align__(16) bf16_t sQ[128 * 64];
  __shared__ __align__(16) bf16_t sK[128 * 64];   // [k-pos][d] == B^T [n][k]
  const int tid = threadIdx.x, lane = tid & 31, w = tid >> 5;
  const int wrow = w >> 2, wcol = w & 3;
  const int q0 = blockIdx.x * 128, k0 = blockIdx.y * 128;
  const int z = blockIdx.z;                        // z = h*B + b
  const int h = z / B_SZ, b = z % B_SZ;
  const size_t base = (size_t)b * LSEQ * DMODEL + (size_t)h * DK;

  if (w == 0) {
    tdm_load_2d_bf16(lds_addr_of(sQ), QH + base + (size_t)q0 * DMODEL,
                     DK, 128, DMODEL, DK, 128);
    tdm_load_2d_bf16(lds_addr_of(sK), KH + base + (size_t)k0 * DMODEL,
                     DK, 128, DMODEL, DK, 128);
    WAIT_TENSOR(0);
  }
  __syncthreads();

  v8f acc[4][2];
  const v8f vzero = {0.f, 0.f, 0.f, 0.f, 0.f, 0.f, 0.f, 0.f};
  for (int i = 0; i < 4; i++)
    for (int j = 0; j < 2; j++) acc[i][j] = vzero;

  const bf16_t* aB = sQ + (wrow * 64 + (lane & 15)) * 64 + ((lane >> 4) << 3);
  const bf16_t* bB = sK + (wcol * 32 + (lane & 15)) * 64 + ((lane >> 4) << 3);
  for (int ks = 0; ks < DK; ks += 32) {
    v16bf b0 = frag_b(bB + ks);
    v16bf b1 = frag_b(bB + 16 * 64 + ks);
    for (int i = 0; i < 4; i++) {
      v16bf a = frag_a(aB + i * 16 * 64 + ks);
      acc[i][0] = WMMA_BF16(a, b0, acc[i][0]);
      acc[i][1] = WMMA_BF16(a, b1, acc[i][1]);
    }
  }

  float* out = P + (size_t)z * LSEQ * LSEQ;
  const int mrB = 8 * (lane >> 4);
  for (int i = 0; i < 4; i++)
    for (int j = 0; j < 2; j++) {
      int n = k0 + wcol * 32 + j * 16 + (lane & 15);
      for (int r = 0; r < 8; r++) {
        int m = q0 + wrow * 64 + i * 16 + mrB + r;
        out[(size_t)m * LSEQ + n] = acc[i][j][r];
      }
    }
}

__global__ __launch_bounds__(256) void softmax_kernel(float* __restrict__ P) {
  __shared__ float red[256];
  const int tid = threadIdx.x;
  float* row = P + (size_t)blockIdx.x * LSEQ;
  float v[4];
  float lmax = -3.402823466e38f;
  for (int j = 0; j < 4; j++) {
    v[j] = row[tid + (j << 8)];
    lmax = fmaxf(lmax, v[j]);
  }
  red[tid] = lmax; __syncthreads();
  for (int s = 128; s; s >>= 1) {
    if (tid < s) red[tid] = fmaxf(red[tid], red[tid + s]);
    __syncthreads();
  }
  const float rm = red[0];
  __syncthreads();
  float lsum = 0.f;
  for (int j = 0; j < 4; j++) { v[j] = __expf(v[j] - rm); lsum += v[j]; }
  red[tid] = lsum; __syncthreads();
  for (int s = 128; s; s >>= 1) {
    if (tid < s) red[tid] += red[tid + s];
    __syncthreads();
  }
  const float inv = 1.0f / red[0];
  for (int j = 0; j < 4; j++) row[tid + (j << 8)] = v[j] * inv;
}

// ctx[b, q, h*64+n] = sum_k P[h*B+b, q, k] * vhT[b, h, n, k]
__global__ __launch_bounds__(256) void attn_v_kernel(
    const float* __restrict__ P, const bf16_t* __restrict__ VHT,
    bf16_t* __restrict__ CTX) {
  __shared__ __align__(16) bf16_t sP[128 * 32];
  __shared__ __align__(16) bf16_t sV[64 * 32];    // [n][k]
  const int tid = threadIdx.x, lane = tid & 31, w = tid >> 5;
  const int wrow = w >> 1, wcol = w & 1;          // 4x2 waves, 32x32 each
  const int q0 = blockIdx.x * 128;
  const int z = blockIdx.y, h = z / B_SZ, b = z % B_SZ;
  const float* Prow = P + (size_t)z * LSEQ * LSEQ;
  const bf16_t* vt = VHT + ((size_t)b * NHEAD + h) * DK * LSEQ;

  v8f acc[2][2];
  const v8f vzero = {0.f, 0.f, 0.f, 0.f, 0.f, 0.f, 0.f, 0.f};
  for (int i = 0; i < 2; i++)
    for (int j = 0; j < 2; j++) acc[i][j] = vzero;

  for (int k0 = 0; k0 < LSEQ; k0 += 32) {
    __syncthreads();
    if (w == 0)   // V tile 64x32 via TDM while all waves stage P
      tdm_load_2d_bf16(lds_addr_of(sV), vt + k0, LSEQ - k0, DK, LSEQ, 32, 64);
    // P tile 128x32: f32 -> bf16, batched loads then packed b128 stores
    {
      const int row = tid >> 1, half = (tid & 1) * 16;   // 16 f32 per thread
      float4 f[4];
      for (int t = 0; t < 4; t++)
        f[t] = *(const float4*)&Prow[(size_t)(q0 + row) * LSEQ + k0 + half + t * 4];
      bf16_t pk[16];
      for (int t = 0; t < 4; t++) {
        pk[t * 4 + 0] = f2bf(f[t].x); pk[t * 4 + 1] = f2bf(f[t].y);
        pk[t * 4 + 2] = f2bf(f[t].z); pk[t * 4 + 3] = f2bf(f[t].w);
      }
      *(uint4*)&sP[row * 32 + half]     = *(const uint4*)&pk[0];
      *(uint4*)&sP[row * 32 + half + 8] = *(const uint4*)&pk[8];
    }
    if (w == 0) WAIT_TENSOR(0);
    __syncthreads();

    const bf16_t* aB = sP + (wrow * 32 + (lane & 15)) * 32 + ((lane >> 4) << 3);
    const bf16_t* bB = sV + (wcol * 32 + (lane & 15)) * 32 + ((lane >> 4) << 3);
    v16bf b0 = frag_b(bB);
    v16bf b1 = frag_b(bB + 16 * 32);
    for (int i = 0; i < 2; i++) {
      v16bf a = frag_a(aB + i * 16 * 32);
      acc[i][0] = WMMA_BF16(a, b0, acc[i][0]);
      acc[i][1] = WMMA_BF16(a, b1, acc[i][1]);
    }
  }

  const int mrB = 8 * (lane >> 4);
  for (int i = 0; i < 2; i++)
    for (int j = 0; j < 2; j++) {
      int n = wcol * 32 + j * 16 + (lane & 15);
      for (int r = 0; r < 8; r++) {
        int m = q0 + wrow * 32 + i * 16 + mrB + r;
        CTX[((size_t)b * LSEQ + m) * DMODEL + (size_t)h * DK + n] =
            f2bf(acc[i][j][r]);
      }
    }
}

extern "C" void kernel_launch(void* const* d_in, const int* in_sizes, int n_in,
                              void* d_out, int out_size, void* d_ws, size_t ws_size,
                              hipStream_t stream) {
  const float* q   = (const float*)d_in[0];
  const float* k   = (const float*)d_in[1];
  const float* v   = (const float*)d_in[2];
  const float* Wq  = (const float*)d_in[3];
  const float* bq  = (const float*)d_in[4];
  const float* Wk  = (const float*)d_in[5];
  const float* bk  = (const float*)d_in[6];
  const float* Wv  = (const float*)d_in[7];
  const float* bv  = (const float*)d_in[8];
  const float* Wfc = (const float*)d_in[9];
  const float* bfc = (const float*)d_in[10];

  float* out  = (float*)d_out;
  float* attn = out + (size_t)MROWS * DMODEL;   // [128, 1024, 1024] f32

  char* ws = (char*)d_ws;
  const size_t WSZ = (size_t)DMODEL * DMODEL * sizeof(bf16_t);   // 2 MB
  const size_t ASZ = (size_t)MROWS * DMODEL * sizeof(bf16_t);    // 16 MB
  bf16_t* wqT  = (bf16_t*)(ws + 0 * WSZ);   // transposed bf16 weights [N][K]
  bf16_t* wkT  = (bf16_t*)(ws + 1 * WSZ);
  bf16_t* wvT  = (bf16_t*)(ws + 2 * WSZ);
  bf16_t* wfcT = (bf16_t*)(ws + 3 * WSZ);
  bf16_t* qB   = (bf16_t*)(ws + 4 * WSZ + 0 * ASZ);
  bf16_t* kB   = (bf16_t*)(ws + 4 * WSZ + 1 * ASZ);
  bf16_t* vB   = (bf16_t*)(ws + 4 * WSZ + 2 * ASZ);
  bf16_t* qh   = (bf16_t*)(ws + 4 * WSZ + 3 * ASZ);
  bf16_t* kh   = (bf16_t*)(ws + 4 * WSZ + 4 * ASZ);
  bf16_t* vhT  = (bf16_t*)(ws + 4 * WSZ + 5 * ASZ);  // [b][h][d][l]
  bf16_t* ctx  = (bf16_t*)(ws + 4 * WSZ + 6 * ASZ);

  const int MA = MROWS * DMODEL;
  dim3 gt(DMODEL / 32, DMODEL / 32);    // (32, 32)
  transpose_cvt_kernel<<<gt, 256, 0, stream>>>(Wq,  wqT);
  transpose_cvt_kernel<<<gt, 256, 0, stream>>>(Wk,  wkT);
  transpose_cvt_kernel<<<gt, 256, 0, stream>>>(Wv,  wvT);
  transpose_cvt_kernel<<<gt, 256, 0, stream>>>(Wfc, wfcT);
  cvt_f32_bf16<<<2048, 256, 0, stream>>>(q, qB, MA);
  cvt_f32_bf16<<<2048, 256, 0, stream>>>(k, kB, MA);
  cvt_f32_bf16<<<2048, 256, 0, stream>>>(v, vB, MA);

  dim3 gp(DMODEL / 128, MROWS / 128);   // (8, 64)
  const float qscale = 0.125f;          // 1/sqrt(64)
  gemm_bf16_kernel<<<gp, 256, 0, stream>>>(qB, wqT, bq, qh, nullptr,
                                           MROWS, DMODEL, DMODEL, qscale, 0);
  gemm_bf16_kernel<<<gp, 256, 0, stream>>>(kB, wkT, bk, kh, nullptr,
                                           MROWS, DMODEL, DMODEL, 1.0f, 0);
  gemm_bf16_kernel<<<gp, 256, 0, stream>>>(vB, wvT, bv, vhT, nullptr,
                                           MROWS, DMODEL, DMODEL, 1.0f, 1);

  dim3 gs(LSEQ / 128, LSEQ / 128, NHEAD * B_SZ);  // (8, 8, 128)
  attn_scores_kernel<<<gs, 256, 0, stream>>>(qh, kh, attn);

  softmax_kernel<<<NHEAD * B_SZ * LSEQ, 256, 0, stream>>>(attn);

  dim3 gav(LSEQ / 128, NHEAD * B_SZ);   // (8, 128)
  attn_v_kernel<<<gav, 256, 0, stream>>>(attn, vhT, ctx);

  gemm_bf16_kernel<<<gp, 256, 0, stream>>>(ctx, wfcT, bfc, nullptr, out,
                                           MROWS, DMODEL, DMODEL, 1.0f, 0);
}
